// MambaBlock_10763188044444
// MI455X (gfx1250) — matrix operand order repaired and checked by reading, not compiled, not run on hardware
//
#include <hip/hip_runtime.h>
#include <hip/hip_bf16.h>
#include <math.h>

// ---------------- problem constants ----------------
#define D_MODEL   1024
#define D_INNER   2048
#define D_STATE   64
#define D_CONV    4
#define HEADDIM   64
#define NHEADS    32
#define CONV_DIM  2176          // D_INNER + 2*D_STATE
#define D_IN_PROJ 4256          // 2*D_INNER + 2*D_STATE + NHEADS
#define BATCH     2
#define SEQLEN    2048
#define NTOK      4096          // BATCH * SEQLEN
#define D_FF      4096
#define EPSV      1e-5f

typedef __attribute__((ext_vector_type(16))) __bf16 bf16x16;
typedef __attribute__((ext_vector_type(8)))  float  v8f;

__device__ __forceinline__ unsigned short f2bf(float f) {
    unsigned int u = __float_as_uint(f);
    unsigned int r = (u + 0x7FFFu + ((u >> 16) & 1u)) >> 16;   // RNE
    return (unsigned short)r;
}

// ---------------- fp32 -> bf16 conversion ----------------
__global__ __launch_bounds__(256) void cvt_f32_bf16_kernel(
    const float* __restrict__ in, unsigned short* __restrict__ out, int n)
{
    int i = blockIdx.x * 256 + threadIdx.x;
    if (i < n) out[i] = f2bf(in[i]);
}

// ---------------- LayerNorm -> bf16 ----------------
__global__ __launch_bounds__(256) void layernorm_bf16_kernel(
    const float* __restrict__ x, const float* __restrict__ w,
    const float* __restrict__ b, unsigned short* __restrict__ out)
{
    int tok = blockIdx.x, tid = threadIdx.x;
    const float* xr = x + (size_t)tok * D_MODEL;
    float v[4]; float s = 0.f;
#pragma unroll
    for (int i = 0; i < 4; ++i) { v[i] = xr[tid + 256 * i]; s += v[i]; }
    __shared__ float red[256];
    red[tid] = s; __syncthreads();
    for (int o = 128; o > 0; o >>= 1) { if (tid < o) red[tid] += red[tid + o]; __syncthreads(); }
    float mu = red[0] * (1.0f / D_MODEL);
    __syncthreads();
    float s2 = 0.f;
#pragma unroll
    for (int i = 0; i < 4; ++i) { float d = v[i] - mu; s2 += d * d; }
    red[tid] = s2; __syncthreads();
    for (int o = 128; o > 0; o >>= 1) { if (tid < o) red[tid] += red[tid + o]; __syncthreads(); }
    float inv = rsqrtf(red[0] * (1.0f / D_MODEL) + EPSV);
#pragma unroll
    for (int i = 0; i < 4; ++i) {
        int c = tid + 256 * i;
        out[(size_t)tok * D_MODEL + c] = f2bf((v[i] - mu) * inv * w[c] + b[c]);
    }
}

// ---------------- WMMA bf16 GEMM:  out[M][N] = A[M][K] * B[N][K]^T ----------------
// EPI: 0 = plain fp32 store
//      1 = fp32 store + residual add
//      2 = +bias, exact GELU, bf16 store
//      3 = +bias + residual, fp32 store
template<int EPI>
__global__ __launch_bounds__(256) void gemm_wmma_bf16_kernel(
    const unsigned short* __restrict__ A,     // M x K bf16 (row major)
    const unsigned short* __restrict__ B,     // N x K bf16 (row major)
    int M, int N, int K,
    const float* __restrict__ bias,           // [N] or null
    const float* __restrict__ res,            // M x N fp32 or null
    float* __restrict__ outF,                 // M x N fp32 or null
    unsigned short* __restrict__ outBF)       // M x N bf16 or null
{
    const int lane = threadIdx.x & 31;
    const int wid  = threadIdx.x >> 5;          // 8 waves
    const int wm   = wid & 1;                   // 2 waves along M
    const int wn   = wid >> 1;                  // 4 waves along N
    const int m_w  = blockIdx.y * 128 + wm * 64;
    const int n_w  = blockIdx.x * 128 + wn * 32;

    const int rsel = lane >> 4;                 // 0: lanes 0-15, 1: lanes 16-31
    const int ridx = lane & 15;

    v8f zero;
#pragma unroll
    for (int e = 0; e < 8; ++e) zero[e] = 0.0f;
    v8f acc[4][2];
#pragma unroll
    for (int i = 0; i < 4; ++i)
#pragma unroll
        for (int j = 0; j < 2; ++j) acc[i][j] = zero;

    // A fragment: lane half 0 holds K=0..7 & 16..23; lane half 1 holds K=8..15 & 24..31
    const unsigned short* aP[4];
#pragma unroll
    for (int i = 0; i < 4; ++i)
        aP[i] = A + (size_t)(m_w + 16 * i + ridx) * K + rsel * 8;
    // B fragment: lane half 0 holds K=0..7 & 16..23; lane half 1 holds K=8..15 & 24..31
    const unsigned short* bP[2];
#pragma unroll
    for (int j = 0; j < 2; ++j) {
        int r = n_w + 16 * j + ridx;
        if (r >= N) r = N - 1;                  // clamp (stores masked later)
        bP[j] = B + (size_t)r * K + rsel * 8;
    }

    union FragU { bf16x16 v; uint4 u[2]; };

    for (int k0 = 0; k0 < K; k0 += 32) {
        FragU a[4], b[2];
#pragma unroll
        for (int i = 0; i < 4; ++i) {
            a[i].u[0] = *(const uint4*)(aP[i] + k0);
            a[i].u[1] = *(const uint4*)(aP[i] + k0 + 16);
        }
#pragma unroll
        for (int j = 0; j < 2; ++j) {
            b[j].u[0] = *(const uint4*)(bP[j] + k0);
            b[j].u[1] = *(const uint4*)(bP[j] + k0 + 16);
            __builtin_prefetch(bP[j] + k0 + 128, 0, 1);   // stream next weight lines
        }
#pragma unroll
        for (int i = 0; i < 4; ++i)
#pragma unroll
            for (int j = 0; j < 2; ++j)
                acc[i][j] = __builtin_amdgcn_wmma_f32_16x16x32_bf16(
                    false, a[i].v, false, b[j].v, (short)0, acc[i][j], false, false);
    }

    // C/D layout: VGPR e, lanes 0-15 -> (M=e, N=lane); lanes 16-31 -> (M=8+e, N=lane-16)
    const int rbase = rsel * 8;
#pragma unroll
    for (int j = 0; j < 2; ++j) {
        int n = n_w + 16 * j + ridx;
        if (n >= N) continue;
        float bv = (EPI == 2 || EPI == 3) ? bias[n] : 0.0f;
#pragma unroll
        for (int i = 0; i < 4; ++i) {
#pragma unroll
            for (int e = 0; e < 8; ++e) {
                int m = m_w + 16 * i + rbase + e;
                float v = acc[i][j][e];
                size_t o = (size_t)m * N + n;
                if (EPI == 0) {
                    outF[o] = v;
                } else if (EPI == 1) {
                    outF[o] = v + res[o];
                } else if (EPI == 2) {
                    float t = v + bv;
                    t = 0.5f * t * (1.0f + erff(t * 0.70710678118654752f));
                    outBF[o] = f2bf(t);
                } else {
                    outF[o] = v + bv + res[o];
                }
            }
        }
    }
}

// ---------------- causal depthwise conv + SiLU ----------------
__global__ __launch_bounds__(256) void conv_silu_kernel(
    const float* __restrict__ zxbcdt, const float* __restrict__ cw,
    const float* __restrict__ cb, float* __restrict__ xBC)
{
    int idx = blockIdx.x * 256 + threadIdx.x;
    if (idx >= NTOK * CONV_DIM) return;
    int tok = idx / CONV_DIM, c = idx - tok * CONV_DIM;
    int l = tok & (SEQLEN - 1);
    int bb = tok >> 11;
    float acc = cb[c];
#pragma unroll
    for (int k = 0; k < D_CONV; ++k) {
        int ls = l + k - (D_CONV - 1);
        if (ls >= 0)
            acc += zxbcdt[(size_t)(bb * SEQLEN + ls) * D_IN_PROJ + D_INNER + c] * cw[c * D_CONV + k];
    }
    acc = acc / (1.0f + expf(-acc));       // SiLU
    xBC[(size_t)tok * CONV_DIM + c] = acc;
}

// ---------------- dt = softplus(raw + dt_bias) ----------------
__global__ __launch_bounds__(256) void dt_kernel(
    const float* __restrict__ zxbcdt, const float* __restrict__ dt_bias,
    float* __restrict__ dt)
{
    int idx = blockIdx.x * 256 + threadIdx.x;
    if (idx >= NTOK * NHEADS) return;
    int tok = idx >> 5, h = idx & 31;
    float x = zxbcdt[(size_t)tok * D_IN_PROJ + (D_INNER + CONV_DIM) + h] + dt_bias[h];
    dt[idx] = (x > 20.f) ? x : log1pf(expf(x));
}

// ---------------- sequential SSM scan: one block per (batch, head) ----------------
__global__ __launch_bounds__(256) void ssm_scan_kernel(
    const float* __restrict__ xBC,   // [NTOK][CONV_DIM]
    const float* __restrict__ dtp,   // [NTOK][NHEADS]
    const float* __restrict__ A_log, const float* __restrict__ Dp,
    float* __restrict__ y)           // [NTOK][D_INNER]
{
    int bh = blockIdx.x;
    int bb = bh >> 5, h = bh & 31;
    int tid = threadIdx.x;
    int p  = tid & 63;               // headdim index
    int g  = tid >> 6;               // 4 groups over state dim
    int n0 = g * 16;

    float Ah = -expf(A_log[h]);
    float Dh = Dp[h];
    float s[16];
#pragma unroll
    for (int j = 0; j < 16; ++j) s[j] = 0.f;

    __shared__ float xs_sh[64], B_sh[64], C_sh[64];
    __shared__ float dt_sh;
    __shared__ float part[4][64];

    const float* base = xBC + (size_t)bb * SEQLEN * CONV_DIM;
    for (int t = 0; t < SEQLEN; ++t) {
        const float* row = base + (size_t)t * CONV_DIM;
        if (tid < 64)        xs_sh[tid]       = row[h * HEADDIM + tid];
        else if (tid < 128)  B_sh[tid - 64]   = row[D_INNER + (tid - 64)];
        else if (tid < 192)  C_sh[tid - 128]  = row[D_INNER + D_STATE + (tid - 128)];
        else if (tid == 192) dt_sh            = dtp[((size_t)bb * SEQLEN + t) * NHEADS + h];
        __syncthreads();
        float dt = dt_sh;
        float dA = expf(dt * Ah);
        float xp = dt * xs_sh[p];
        float acc = 0.f;
#pragma unroll
        for (int j = 0; j < 16; ++j) {
            s[j] = s[j] * dA + xp * B_sh[n0 + j];
            acc += s[j] * C_sh[n0 + j];
        }
        part[g][p] = acc;
        __syncthreads();
        if (tid < 64) {
            float r = part[0][tid] + part[1][tid] + part[2][tid] + part[3][tid]
                    + Dh * xs_sh[tid];
            y[((size_t)bb * SEQLEN + t) * D_INNER + h * HEADDIM + tid] = r;
        }
        __syncthreads();
    }
}

// ---------------- gate (y * silu(z)) + RMSNorm -> bf16 ----------------
__global__ __launch_bounds__(256) void gate_rmsnorm_kernel(
    const float* __restrict__ y, const float* __restrict__ zxbcdt,
    const float* __restrict__ norm_w, unsigned short* __restrict__ out)
{
    int tok = blockIdx.x, tid = threadIdx.x;
    const float* yr = y + (size_t)tok * D_INNER;
    const float* zr = zxbcdt + (size_t)tok * D_IN_PROJ;   // z = first D_INNER cols
    float v[8]; float ss = 0.f;
#pragma unroll
    for (int i = 0; i < 8; ++i) {
        int c = tid + i * 256;
        float z = zr[c];
        float t = yr[c] * (z / (1.f + expf(-z)));
        v[i] = t; ss += t * t;
    }
    __shared__ float red[256];
    red[tid] = ss; __syncthreads();
    for (int o = 128; o > 0; o >>= 1) { if (tid < o) red[tid] += red[tid + o]; __syncthreads(); }
    float scale = rsqrtf(red[0] * (1.0f / D_INNER) + EPSV);
#pragma unroll
    for (int i = 0; i < 8; ++i) {
        int c = tid + i * 256;
        out[(size_t)tok * D_INNER + c] = f2bf(v[i] * scale * norm_w[c]);
    }
}

// ---------------- host side ----------------
static inline size_t alignup256(size_t v) { return (v + 255) & ~(size_t)255; }

extern "C" void kernel_launch(void* const* d_in, const int* in_sizes, int n_in,
                              void* d_out, int out_size, void* d_ws, size_t ws_size,
                              hipStream_t stream)
{
    (void)in_sizes; (void)n_in; (void)out_size; (void)ws_size;

    const float* x         = (const float*)d_in[0];
    const float* ln1_w     = (const float*)d_in[1];
    const float* ln1_b     = (const float*)d_in[2];
    const float* in_proj_w = (const float*)d_in[3];
    const float* conv_w    = (const float*)d_in[4];
    const float* conv_b    = (const float*)d_in[5];
    const float* dt_bias   = (const float*)d_in[6];
    const float* A_log     = (const float*)d_in[7];
    const float* Dparam    = (const float*)d_in[8];
    const float* norm_w    = (const float*)d_in[9];
    const float* out_proj_w= (const float*)d_in[10];
    const float* ln2_w     = (const float*)d_in[11];
    const float* ln2_b     = (const float*)d_in[12];
    const float* mlp_w1    = (const float*)d_in[13];
    const float* mlp_b1    = (const float*)d_in[14];
    const float* mlp_w2    = (const float*)d_in[15];
    const float* mlp_b2    = (const float*)d_in[16];
    float* out             = (float*)d_out;

    unsigned char* ws = (unsigned char*)d_ws;
    size_t off = 0;
    auto take = [&](size_t bytes) -> void* {
        void* p = ws + off; off = alignup256(off + bytes); return p;
    };

    unsigned short* Wip_bf  = (unsigned short*)take((size_t)D_IN_PROJ * D_MODEL * 2);
    unsigned short* Wop_bf  = (unsigned short*)take((size_t)D_MODEL * D_INNER * 2);
    unsigned short* Wm1_bf  = (unsigned short*)take((size_t)D_FF * D_MODEL * 2);
    unsigned short* Wm2_bf  = (unsigned short*)take((size_t)D_MODEL * D_FF * 2);
    unsigned short* h1_bf   = (unsigned short*)take((size_t)NTOK * D_MODEL * 2);
    float*          zxbcdt  = (float*)take((size_t)NTOK * D_IN_PROJ * 4);
    float*          xBC     = (float*)take((size_t)NTOK * CONV_DIM * 4);
    float*          dt_ws   = (float*)take((size_t)NTOK * NHEADS * 4);
    float*          y_ws    = (float*)take((size_t)NTOK * D_INNER * 4);
    unsigned short* ybf     = (unsigned short*)take((size_t)NTOK * D_INNER * 2);
    float*          x2      = (float*)take((size_t)NTOK * D_MODEL * 4);
    unsigned short* h2_bf   = (unsigned short*)take((size_t)NTOK * D_MODEL * 2);
    unsigned short* mid_bf  = (unsigned short*)take((size_t)NTOK * D_FF * 2);

    // 1) weights -> bf16 (once per launch; deterministic)
    {
        int n;
        n = D_IN_PROJ * D_MODEL;
        cvt_f32_bf16_kernel<<<(n + 255) / 256, 256, 0, stream>>>(in_proj_w, Wip_bf, n);
        n = D_MODEL * D_INNER;
        cvt_f32_bf16_kernel<<<(n + 255) / 256, 256, 0, stream>>>(out_proj_w, Wop_bf, n);
        n = D_FF * D_MODEL;
        cvt_f32_bf16_kernel<<<(n + 255) / 256, 256, 0, stream>>>(mlp_w1, Wm1_bf, n);
        n = D_MODEL * D_FF;
        cvt_f32_bf16_kernel<<<(n + 255) / 256, 256, 0, stream>>>(mlp_w2, Wm2_bf, n);
    }

    // 2) LN1 -> bf16
    layernorm_bf16_kernel<<<NTOK, 256, 0, stream>>>(x, ln1_w, ln1_b, h1_bf);

    // 3) in_proj: [4096 x 1024] x [4256 x 1024]^T -> zxbcdt fp32
    {
        dim3 grid((D_IN_PROJ + 127) / 128, NTOK / 128);
        gemm_wmma_bf16_kernel<0><<<grid, 256, 0, stream>>>(
            h1_bf, Wip_bf, NTOK, D_IN_PROJ, D_MODEL, nullptr, nullptr, zxbcdt, nullptr);
    }

    // 4) conv + SiLU
    {
        int n = NTOK * CONV_DIM;
        conv_silu_kernel<<<(n + 255) / 256, 256, 0, stream>>>(zxbcdt, conv_w, conv_b, xBC);
    }

    // 5) dt = softplus
    {
        int n = NTOK * NHEADS;
        dt_kernel<<<(n + 255) / 256, 256, 0, stream>>>(zxbcdt, dt_bias, dt_ws);
    }

    // 6) sequential SSM scan
    ssm_scan_kernel<<<BATCH * NHEADS, 256, 0, stream>>>(xBC, dt_ws, A_log, Dparam, y_ws);

    // 7) gate + RMSNorm -> bf16
    gate_rmsnorm_kernel<<<NTOK, 256, 0, stream>>>(y_ws, zxbcdt, norm_w, ybf);

    // 8) out_proj + residual(x) -> x2 fp32
    {
        dim3 grid(D_MODEL / 128, NTOK / 128);
        gemm_wmma_bf16_kernel<1><<<grid, 256, 0, stream>>>(
            ybf, Wop_bf, NTOK, D_MODEL, D_INNER, nullptr, x, x2, nullptr);
    }

    // 9) LN2 -> bf16
    layernorm_bf16_kernel<<<NTOK, 256, 0, stream>>>(x2, ln2_w, ln2_b, h2_bf);

    // 10) MLP1 + bias + GELU -> bf16
    {
        dim3 grid(D_FF / 128, NTOK / 128);
        gemm_wmma_bf16_kernel<2><<<grid, 256, 0, stream>>>(
            h2_bf, Wm1_bf, NTOK, D_FF, D_MODEL, mlp_b1, nullptr, nullptr, mid_bf);
    }

    // 11) MLP2 + bias + residual(x2) -> out fp32
    {
        dim3 grid(D_MODEL / 128, NTOK / 128);
        gemm_wmma_bf16_kernel<3><<<grid, 256, 0, stream>>>(
            mid_bf, Wm2_bf, NTOK, D_MODEL, D_FF, mlp_b2, x2, out, nullptr);
    }
}
